// s4ormer_63247688401500
// MI455X (gfx1250) — compile-verified, matrix operand order checked by reading, not run on hardware
//
#include <hip/hip_runtime.h>

// ============================================================================
// s4ormer on MI455X (gfx1250, wave32, WMMA).
//  - FFT conv replaced by the exact S4D recurrence (VALU scan, modes split 4x).
//  - All GEMMs via v_wmma_f32_16x16x32_bf16 with fused GLU/SwiGLU/bias/residual
//    epilogues (avoids the 512MB FF intermediate entirely).
//  - Double-buffered LDS pipeline using GLOBAL_LOAD_ASYNC_TO_LDS_B128 +
//    s_wait_asynccnt (builtin confirmed present by round-2 diagnostics).
//  - Workspace use: ~354 MB.
// ============================================================================

constexpr int D_DEPTH = 4, D_S4 = 2, D_DIM = 512, D_N2 = 32, D_FF = 2048;
constexpr int D_B = 16, D_L = 2048;
constexpr long NTOK = (long)D_B * D_L;   // 32768 rows

typedef __attribute__((ext_vector_type(8)))  float  v8f;
typedef __attribute__((ext_vector_type(16))) __bf16 v16bf;
typedef __attribute__((ext_vector_type(4)))  int    v4i;

#if defined(__has_builtin)
#if __has_builtin(__builtin_amdgcn_global_load_async_to_lds_b128)
#define HAVE_ASYNC_LDS 1
#endif
#endif
#ifndef HAVE_ASYNC_LDS
#define HAVE_ASYNC_LDS 0
#endif

typedef __attribute__((address_space(1))) v4i* as1_v4ip;
typedef __attribute__((address_space(3))) v4i* as3_v4ip;

__device__ __forceinline__ void async_copy16(const unsigned short* g,
                                             unsigned short* l) {
#if HAVE_ASYNC_LDS
  __builtin_amdgcn_global_load_async_to_lds_b128((as1_v4ip)g, (as3_v4ip)l, 0,
                                                 0);
#else
  uint4 v = *(const uint4*)g;
  *(uint4*)l = v;
#endif
}

template <int N>
__device__ __forceinline__ void wait_async() {
#if HAVE_ASYNC_LDS
#if __has_builtin(__builtin_amdgcn_s_wait_asynccnt)
  __builtin_amdgcn_s_wait_asynccnt(N);
#else
  asm volatile("s_wait_asynccnt %0" ::"i"(N) : "memory");
#endif
#endif
}

__device__ __forceinline__ unsigned short f2bf(float f) {
  union { float f; unsigned int u; } v; v.f = f;
  unsigned int r = v.u + 0x7fffu + ((v.u >> 16) & 1u);   // round-nearest-even
  return (unsigned short)(r >> 16);
}

// ---------------------------------------------------------------- f32 -> bf16
__global__ void cvt_bf16_kernel(const float* __restrict__ s,
                                unsigned short* __restrict__ d, long n) {
  long i  = (long)blockIdx.x * blockDim.x + threadIdx.x;
  long st = (long)gridDim.x * blockDim.x;
  for (; i < n; i += st) d[i] = f2bf(s[i]);
}

// ---------------------------------------------------------------- LayerNorm
__global__ __launch_bounds__(128) void ln_kernel(
    const float* __restrict__ x, const float* __restrict__ w,
    const float* __restrict__ b, float* __restrict__ oF,
    unsigned short* __restrict__ oB, int writeB) {
  __shared__ float rs[128], rq[128];
  long row = blockIdx.x;
  int t = threadIdx.x;
  float4 v = ((const float4*)(x + row * D_DIM))[t];
  rs[t] = v.x + v.y + v.z + v.w;
  rq[t] = v.x * v.x + v.y * v.y + v.z * v.z + v.w * v.w;
  __syncthreads();
  for (int s = 64; s > 0; s >>= 1) {
    if (t < s) { rs[t] += rs[t + s]; rq[t] += rq[t + s]; }
    __syncthreads();
  }
  float mu  = rs[0] * (1.f / D_DIM);
  float inv = rsqrtf(rq[0] * (1.f / D_DIM) - mu * mu + 1e-5f);
  int c = t * 4;
  float4 o;
  o.x = (v.x - mu) * inv * w[c + 0] + b[c + 0];
  o.y = (v.y - mu) * inv * w[c + 1] + b[c + 1];
  o.z = (v.z - mu) * inv * w[c + 2] + b[c + 2];
  o.w = (v.w - mu) * inv * w[c + 3] + b[c + 3];
  ((float4*)(oF + row * D_DIM))[t] = o;
  if (writeB) {
    unsigned short* p = oB + row * D_DIM + c;
    p[0] = f2bf(o.x); p[1] = f2bf(o.y); p[2] = f2bf(o.z); p[3] = f2bf(o.w);
  }
}

// ------------------------------------------------- S4D recurrence (= FFT conv)
__global__ __launch_bounds__(256) void s4_scan_kernel(
    const float* __restrict__ u,
    const float* __restrict__ logA_re, const float* __restrict__ A_im,
    const float* __restrict__ C_re,    const float* __restrict__ C_im,
    const float* __restrict__ log_dt,  const float* __restrict__ Dv,
    float* __restrict__ yF, unsigned short* __restrict__ yB) {
  constexpr int NM = 8;
  int tid = blockIdx.x * blockDim.x + threadIdx.x;  // B*DIM*4 threads
  int q  = tid & 3;
  int bh = tid >> 2;
  int h  = bh & (D_DIM - 1);
  int b  = bh >> 9;
  float dt = __expf(log_dt[h]);
  float wr[NM], wi[NM], cr[NM], ci[NM], sr[NM], si[NM];
#pragma unroll
  for (int m = 0; m < NM; m++) {
    int idx = h * D_N2 + (q * NM + m);
    float Are = -__expf(logA_re[idx]);
    float Aim = A_im[idx];
    float e   = __expf(Are * dt);
    float wre = e * __cosf(Aim * dt);
    float wie = e * __sinf(Aim * dt);
    float nr = wre - 1.f, ni = wie;
    float den = 1.f / (Are * Are + Aim * Aim);
    float qr = (nr * Are + ni * Aim) * den;          // (w-1)/A
    float qi = (ni * Are - nr * Aim) * den;
    float Cr = C_re[idx], Ci = C_im[idx];
    cr[m] = 2.f * (Cr * qr - Ci * qi);               // 2*C*(w-1)/A
    ci[m] = 2.f * (Cr * qi + Ci * qr);
    wr[m] = wre; wi[m] = wie; sr[m] = 0.f; si[m] = 0.f;
  }
  float Dh = Dv[h];
  long base = ((long)b * D_L) * D_DIM + h;
  for (int l = 0; l < D_L; l++) {
    float uv = u[base + (long)l * D_DIM];
    float p = 0.f;
#pragma unroll
    for (int m = 0; m < NM; m++) {
      float nsr = wr[m] * sr[m] - wi[m] * si[m] + uv;
      float nsi = wr[m] * si[m] + wi[m] * sr[m];
      sr[m] = nsr; si[m] = nsi;
      p += cr[m] * nsr - ci[m] * nsi;                // Re(C'*s)
    }
    p += __shfl_xor(p, 1, 32);
    p += __shfl_xor(p, 2, 32);
    if (q == 0) {
      float yv = p + uv * Dh;
      float targ = yv * (0.7978845608f * (1.f + 0.044715f * yv * yv));
      float gl = 0.5f * yv * (1.f + tanhf(targ));    // tanh-approx GELU
      long o = base + (long)l * D_DIM;
      yF[o] = gl;
      yB[o] = f2bf(gl);
    }
  }
}

// ---------------------------------------------------------------- WMMA GEMM
// C = act(X@Wa^T + ba, X@Wg^T + bg) [+res]; MODE 0: a*sigmoid(g),
// MODE 1: a*silu(g), MODE 2: plain a + ba. X: MxK bf16, W: NxK bf16.
// LK=40 bf16 pad: 80B row stride -> 16B-aligned B128 chunks, 20-bank stride
// (20*r mod 64 distinct for r=0..15 -> conflict-free fragment gathers).
constexpr int BM = 128, BN = 64, BK = 32, LK = 40;

template <int MODE>
__device__ __forceinline__ void load_stage(
    unsigned short* sAb, unsigned short* sWab, unsigned short* sWgb,
    const unsigned short* __restrict__ X, const unsigned short* __restrict__ Wa,
    const unsigned short* __restrict__ Wg, int bm, int bn, int K, int k0,
    int tid) {
  {  // A tile: 128x32 bf16 = 512 chunks of 16B; 2 per thread (same row)
    int e = tid * 2;
    int row = e >> 2, cu = (e & 3) * 4;              // dword col {0,4,8,12}
    const unsigned short* g = X + (size_t)(bm + row) * K + k0 + cu * 2;
    unsigned short* l = sAb + row * LK + cu * 2;
#if !HAVE_ASYNC_LDS
    if (k0 + BK < K) __builtin_prefetch(g + BK, 0, 0);
#endif
    async_copy16(g, l);
    async_copy16(g + 8, l + 8);
  }
  {  // W tiles: 64x32 bf16 = 256 chunks; 1 per thread (each)
    int row = tid >> 2, cu = (tid & 3) * 4;
    async_copy16(Wa + (size_t)(bn + row) * K + k0 + cu * 2,
                 sWab + row * LK + cu * 2);
    if constexpr (MODE < 2)
      async_copy16(Wg + (size_t)(bn + row) * K + k0 + cu * 2,
                   sWgb + row * LK + cu * 2);
  }
}

template <int MODE>
__global__ __launch_bounds__(256) void wmma_gemm_kernel(
    const unsigned short* __restrict__ X,
    const unsigned short* __restrict__ Wa,
    const unsigned short* __restrict__ Wg,
    const float* __restrict__ ba, const float* __restrict__ bg,
    const float* __restrict__ res,
    float* __restrict__ oF, unsigned short* __restrict__ oB,
    int N, int K) {
  __shared__ unsigned short sA[2][BM * LK];
  __shared__ unsigned short sWa[2][BN * LK];
  __shared__ unsigned short sWg[2][(MODE < 2) ? BN * LK : 1];
  const int bm = blockIdx.x * BM, bn = blockIdx.y * BN;
  const int tid = threadIdx.x, lane = tid & 31, wave = tid >> 5;
  const int wm = wave >> 1, wn = wave & 1;           // 4x2 waves, 32x32 each
  v8f z;
#pragma unroll
  for (int i = 0; i < 8; i++) z[i] = 0.f;
  v8f accA[2][2], accG[2][2];
#pragma unroll
  for (int i = 0; i < 2; i++)
#pragma unroll
    for (int j = 0; j < 2; j++) { accA[i][j] = z; accG[i][j] = z; }

  int s = 0;
  load_stage<MODE>(sA[0], sWa[0], sWg[0], X, Wa, Wg, bm, bn, K, 0, tid);

  for (int k0 = 0; k0 < K; k0 += BK) {
    if (k0 + BK < K) {
      load_stage<MODE>(sA[s ^ 1], sWa[s ^ 1], sWg[s ^ 1], X, Wa, Wg, bm, bn, K,
                       k0 + BK, tid);
      wait_async<(MODE < 2) ? 4 : 3>();   // older stage complete
    } else {
      wait_async<0>();
    }
    __syncthreads();

    // fragments per ISA 7.12.2: row = lane%16, K halves by lane/16
    const int r16 = lane & 15, kb = (lane >> 4) * 4;  // dword K offset
    union U { v16bf v; unsigned int u[8]; };
    U a[2], fA[2], fG[2];
#pragma unroll
    for (int tm = 0; tm < 2; tm++) {
      const unsigned int* p =
          (const unsigned int*)(sA[s] + (wm * 32 + tm * 16 + r16) * LK);
#pragma unroll
      for (int v2 = 0; v2 < 4; v2++) {
        a[tm].u[v2] = p[kb + v2];
        a[tm].u[4 + v2] = p[8 + kb + v2];
      }
    }
#pragma unroll
    for (int tn = 0; tn < 2; tn++) {
      const unsigned int* p =
          (const unsigned int*)(sWa[s] + (wn * 32 + tn * 16 + r16) * LK);
#pragma unroll
      for (int v2 = 0; v2 < 4; v2++) {
        fA[tn].u[v2] = p[kb + v2];
        fA[tn].u[4 + v2] = p[8 + kb + v2];
      }
      if constexpr (MODE < 2) {
        const unsigned int* pg =
            (const unsigned int*)(sWg[s] + (wn * 32 + tn * 16 + r16) * LK);
#pragma unroll
        for (int v2 = 0; v2 < 4; v2++) {
          fG[tn].u[v2] = pg[kb + v2];
          fG[tn].u[4 + v2] = pg[8 + kb + v2];
        }
      }
    }
#pragma unroll
    for (int tm = 0; tm < 2; tm++)
#pragma unroll
      for (int tn = 0; tn < 2; tn++) {
        accA[tm][tn] = __builtin_amdgcn_wmma_f32_16x16x32_bf16(
            false, a[tm].v, false, fA[tn].v, (short)0, accA[tm][tn], false,
            false);
        if constexpr (MODE < 2)
          accG[tm][tn] = __builtin_amdgcn_wmma_f32_16x16x32_bf16(
              false, a[tm].v, false, fG[tn].v, (short)0, accG[tm][tn], false,
              false);
      }
    __syncthreads();   // all waves done reading stage s before it is refilled
    s ^= 1;
  }

  // epilogue: D layout = vgpr r -> M=r (lanes<16) / r+8 (lanes>=16), N=lane%16
#pragma unroll
  for (int tm = 0; tm < 2; tm++)
#pragma unroll
    for (int tn = 0; tn < 2; tn++) {
      const int mB = bm + wm * 32 + tm * 16 + ((lane >> 4) << 3);
      const int nI = bn + wn * 32 + tn * 16 + (lane & 15);
      const float bAv = ba[nI];
      float bGv = 0.f;
      if constexpr (MODE < 2) bGv = bg[nI];
#pragma unroll
      for (int r = 0; r < 8; r++) {
        const size_t oi = (size_t)(mB + r) * N + nI;
        float av = accA[tm][tn][r] + bAv;
        float o;
        if constexpr (MODE == 0) {
          float g = accG[tm][tn][r] + bGv;
          o = av * (1.f / (1.f + __expf(-g)));
        } else if constexpr (MODE == 1) {
          float g = accG[tm][tn][r] + bGv;
          o = av * g * (1.f / (1.f + __expf(-g)));
        } else {
          o = av;
        }
        if (res) o += res[oi];
        if (oF) oF[oi] = o;
        if (oB) oB[oi] = f2bf(o);
      }
    }
}

// ============================================================================
extern "C" void kernel_launch(void* const* d_in, const int* in_sizes, int n_in,
                              void* d_out, int out_size, void* d_ws,
                              size_t ws_size, hipStream_t stream) {
  const float* x_in = (const float*)d_in[0];
  const float* ln1w = (const float*)d_in[1];
  const float* ln1b = (const float*)d_in[2];
  const float* ln2w = (const float*)d_in[3];
  const float* ln2b = (const float*)d_in[4];
  const float* lAre = (const float*)d_in[5];
  const float* Aim  = (const float*)d_in[6];
  const float* Cre  = (const float*)d_in[7];
  const float* Cim  = (const float*)d_in[8];
  const float* ldt  = (const float*)d_in[9];
  const float* Dv   = (const float*)d_in[10];
  const float* s4w  = (const float*)d_in[11];
  const float* s4b  = (const float*)d_in[12];
  const float* w1   = (const float*)d_in[13];
  const float* b1   = (const float*)d_in[14];
  const float* w2   = (const float*)d_in[15];
  const float* b2   = (const float*)d_in[16];
  float* xb = (float*)d_out;  // d_out doubles as the persistent x buffer

  char* p = (char*)d_ws;
  float* hbuf = (float*)p;            p += NTOK * D_DIM * sizeof(float);
  float* ybuf = (float*)p;            p += NTOK * D_DIM * sizeof(float);
  unsigned short* ybf   = (unsigned short*)p; p += NTOK * D_DIM * 2;
  unsigned short* hbf   = (unsigned short*)p; p += NTOK * D_DIM * 2;
  unsigned short* glubf = (unsigned short*)p; p += NTOK * D_FF * 2;
  unsigned short* s4wbf = (unsigned short*)p; p += (long)D_DEPTH * D_S4 * 2 * D_DIM * D_DIM * 2;
  unsigned short* w1bf  = (unsigned short*)p; p += (long)D_DEPTH * 2 * D_FF * D_DIM * 2;
  unsigned short* w2bf  = (unsigned short*)p; p += (long)D_DEPTH * D_DIM * D_FF * 2;

  (void)hipMemcpyAsync(xb, x_in, NTOK * D_DIM * sizeof(float),
                       hipMemcpyDeviceToDevice, stream);

  cvt_bf16_kernel<<<2048, 256, 0, stream>>>(s4w, s4wbf,
      (long)D_DEPTH * D_S4 * 2 * D_DIM * D_DIM);
  cvt_bf16_kernel<<<2048, 256, 0, stream>>>(w1, w1bf,
      (long)D_DEPTH * 2 * D_FF * D_DIM);
  cvt_bf16_kernel<<<2048, 256, 0, stream>>>(w2, w2bf,
      (long)D_DEPTH * D_DIM * D_FF);

  dim3 gS4(NTOK / BM, D_DIM / BN);   // (256, 8)
  dim3 gFF1(NTOK / BM, D_FF / BN);   // (256, 32)
  dim3 gFF2(NTOK / BM, D_DIM / BN);  // (256, 8)

  for (int i = 0; i < D_DEPTH; i++) {
    ln_kernel<<<NTOK, 128, 0, stream>>>(xb, ln1w + i * D_DIM, ln1b + i * D_DIM,
                                        hbuf, hbf, 0);
    for (int j = 0; j < D_S4; j++) {
      int off = i * D_S4 + j;
      s4_scan_kernel<<<(D_B * D_DIM * 4) / 256, 256, 0, stream>>>(
          hbuf, lAre + (long)off * D_DIM * D_N2, Aim + (long)off * D_DIM * D_N2,
          Cre + (long)off * D_DIM * D_N2, Cim + (long)off * D_DIM * D_N2,
          ldt + (long)off * D_DIM, Dv + (long)off * D_DIM, ybuf, ybf);
      const unsigned short* Wa = s4wbf + (long)off * 2 * D_DIM * D_DIM;
      const unsigned short* Wg = Wa + (long)D_DIM * D_DIM;
      const float* bb = s4b + (long)off * 2 * D_DIM;
      if (j + 1 < D_S4)  // h = glu(proj(y))
        wmma_gemm_kernel<0><<<gS4, 256, 0, stream>>>(
            ybf, Wa, Wg, bb, bb + D_DIM, nullptr, hbuf, nullptr, D_DIM, D_DIM);
      else               // x = glu(proj(y)) + x   (residual fused)
        wmma_gemm_kernel<0><<<gS4, 256, 0, stream>>>(
            ybf, Wa, Wg, bb, bb + D_DIM, xb, xb, nullptr, D_DIM, D_DIM);
    }
    ln_kernel<<<NTOK, 128, 0, stream>>>(xb, ln2w + i * D_DIM, ln2b + i * D_DIM,
                                        hbuf, hbf, 1);
    // glu = a * silu(g), written bf16 only (no 512MB z intermediate)
    wmma_gemm_kernel<1><<<gFF1, 256, 0, stream>>>(
        hbf, w1bf + (long)i * 2 * D_FF * D_DIM,
        w1bf + (long)i * 2 * D_FF * D_DIM + (long)D_FF * D_DIM,
        b1 + (long)i * 2 * D_FF, b1 + (long)i * 2 * D_FF + D_FF, nullptr,
        nullptr, glubf, D_FF, D_DIM);
    // x = glu @ W2^T + b2 + x
    wmma_gemm_kernel<2><<<gFF2, 256, 0, stream>>>(
        glubf, w2bf + (long)i * D_DIM * D_FF, nullptr, b2 + (long)i * D_DIM,
        nullptr, xb, xb, nullptr, D_DIM, D_FF);
  }
}
